// MaskedAutoencoderViT_84456236909234
// MI455X (gfx1250) — matrix operand order repaired and checked by reading, not compiled, not run on hardware
//
#include <hip/hip_runtime.h>
#include <hip/hip_bf16.h>
#include <math.h>

// ---------------------------------------------------------------------------
// Types for CDNA5 WMMA
// ---------------------------------------------------------------------------
typedef __attribute__((ext_vector_type(16))) __bf16 v16bf;
typedef __attribute__((ext_vector_type(8)))  float  v8f;
typedef __attribute__((ext_vector_type(4)))  int    v4i;

// ---------------------------------------------------------------------------
// CDNA5 async global->LDS path (ASYNCcnt-tracked), compile-safe.
// Builtin prototype (from hipcc diagnostic): (v4i AS1*, v4i AS3*, imm, imm).
// ---------------------------------------------------------------------------
#if defined(__AMDGCN__) && __has_builtin(__builtin_amdgcn_global_load_async_to_lds_b128)
#define HAVE_ASYNC_LDS 1
#else
#define HAVE_ASYNC_LDS 0
#endif

__device__ __forceinline__ void copy16_to_lds(void* lds_dst, const void* gsrc) {
#if HAVE_ASYNC_LDS
  __builtin_amdgcn_global_load_async_to_lds_b128(
      (__attribute__((address_space(1))) v4i*)gsrc,
      (__attribute__((address_space(3))) v4i*)lds_dst,
      0, 0);
#else
  *(uint4*)lds_dst = *(const uint4*)gsrc;
#endif
}

__device__ __forceinline__ void wait_async_lds() {
#if HAVE_ASYNC_LDS
#if __has_builtin(__builtin_amdgcn_s_wait_asynccnt)
  __builtin_amdgcn_s_wait_asynccnt(0);
#else
  asm volatile("s_wait_asynccnt 0" ::: "memory");
#endif
#endif
}

__device__ __forceinline__ unsigned pack_bf16(float lo, float hi) {
  unsigned short a = __builtin_bit_cast(unsigned short, (__bf16)lo);
  unsigned short b = __builtin_bit_cast(unsigned short, (__bf16)hi);
  return (unsigned)a | ((unsigned)b << 16);
}

// ---------------------------------------------------------------------------
// Model dimensions (match reference)
// ---------------------------------------------------------------------------
constexpr int BB = 2, NT = 1024;
constexpr int DE = 1024, HE = 16, FE = 4096, LE = 8;   // encoder
constexpr int DD = 512,  HD = 4,  FD = 2048, LDN = 4;  // decoder
constexpr int OUTD = 1;
constexpr int MT = BB * NT;                            // 2048 token rows

// ---------------------------------------------------------------------------
// GEMM tiling: 128x128 block tile, K-step 32, 8 wave32 waves per block.
// Each wave owns a 32x64 patch -> 8 v_wmma_f32_16x16x32_bf16 per K-step.
// Double-buffered LDS (ping-pong), one barrier per K-step.
// ---------------------------------------------------------------------------
constexpr int BM = 128, BN = 128, BK = 32;
constexpr int LDSA = BK + 8;   // padded LDS row stride (bf16 elems)

enum { EPI_NONE = 0, EPI_BIAS = 1, EPI_BIAS_GELU = 2, EPI_BIAS_RES = 3 };

__device__ __forceinline__ float gelu_f(float x) {
  return 0.5f * x * (1.0f + erff(x * 0.7071067811865476f));
}

union FragU { v16bf v; unsigned u[8]; };

// D = alpha*(A x B) [+ bias] [gelu] [+ resid]
// A: bf16 row-major (Mdim x Kdim), lda.
// B: if B_TRANS: bf16 row-major (Ndim x Kdim)  (computes A*B^T)
//    else:       row-major (Kdim x Ndim), fp32 if B_F32 else bf16.
// Per-batch base: ptr + (z/zdiv)*so + (z%zdiv)*si  (covers per-(b,head) slices).
template<bool B_F32, bool B_TRANS, int EPI, bool OUT_BF16>
__global__ __launch_bounds__(256) void gemm_wmma_kernel(
    const __bf16* __restrict__ A, long long aso, long long asi, int lda,
    const void*  __restrict__ Bp, long long bso, long long bsi, int ldb,
    void*        __restrict__ Cp, long long cso, long long csi, int ldc,
    const float* __restrict__ bias,
    const float* __restrict__ Rp, long long rso, long long rsi,
    int zdiv, int Mdim, int Ndim, int Kdim, float alpha)
{
  __shared__ __align__(16) __bf16 Asm[2][BM * LDSA];
  __shared__ __align__(16) __bf16 Bsm[2][BN * LDSA];

  const int tid   = threadIdx.x;
  const int wave  = tid >> 5;
  const int lane  = tid & 31;
  const int waveM = wave >> 1, waveN = wave & 1;
  const int lhalf = lane >> 4, lmod = lane & 15;

  const int z  = blockIdx.z;
  const int zq = z / zdiv, zr = z % zdiv;
  const __bf16* Ab  = A + (long long)zq * aso + (long long)zr * asi;
  const float*  BpF = (const float*)Bp  + (long long)zq * bso + (long long)zr * bsi;
  const __bf16* BpH = (const __bf16*)Bp + (long long)zq * bso + (long long)zr * bsi;

  const int rowBase = blockIdx.y * BM;
  const int colBase = blockIdx.x * BN;

  // ---- tile staging helpers -------------------------------------------
  auto stageA = [&](int buf, int k0) {
#pragma unroll
    for (int c = 0; c < 2; ++c) {
      int chunk = tid + c * 256;          // 0..511 -> 128 rows x 4 octets
      int m  = chunk >> 2;
      int kk = (chunk & 3) << 3;          // 0,8,16,24
      int gm = rowBase + m;
      __bf16* dst = &Asm[buf][m * LDSA + kk];
      if (gm < Mdim) copy16_to_lds(dst, Ab + (long long)gm * lda + (k0 + kk));
      else           *(uint4*)dst = make_uint4(0u, 0u, 0u, 0u);
    }
  };
  auto stageB = [&](int buf, int k0) {
    if (B_TRANS) {
      // bf16 N-major source: straight 16B copies into [n][k] layout
#pragma unroll
      for (int c = 0; c < 2; ++c) {
        int chunk = tid + c * 256;        // 0..511 -> 128 n x 4 octets
        int n  = chunk >> 2;
        int kk = (chunk & 3) << 3;
        int gn = colBase + n;
        __bf16* dst = &Bsm[buf][n * LDSA + kk];
        if (gn < Ndim) copy16_to_lds(dst, BpH + (long long)gn * ldb + (k0 + kk));
        else           *(uint4*)dst = make_uint4(0u, 0u, 0u, 0u);
      }
    } else {
      // K-major source (f32 weights or bf16): two rows -> one packed b32 store
#pragma unroll
      for (int i = 0; i < 8; ++i) {
        int w  = tid + i * 256;           // 0..2047 packed 32-bit words
        int n  = w & 127;                 // 0..127
        int jp = w >> 7;                  // 0..15 K-pair
        int gn = colBase + n;
        float lo = 0.0f, hi = 0.0f;
        if (gn < Ndim) {
          if (B_F32) {
            lo = BpF[(long long)(k0 + 2 * jp) * ldb + gn];
            hi = BpF[(long long)(k0 + 2 * jp + 1) * ldb + gn];
          } else {
            lo = (float)BpH[(long long)(k0 + 2 * jp) * ldb + gn];
            hi = (float)BpH[(long long)(k0 + 2 * jp + 1) * ldb + gn];
          }
        }
        *(unsigned*)(&Bsm[buf][n * LDSA + 2 * jp]) = pack_bf16(lo, hi);
      }
    }
  };

  v8f acc[2][4] = {};
  const int nk = (Kdim + BK - 1) / BK;

  stageA(0, 0);
  stageB(0, 0);
  wait_async_lds();
  __syncthreads();

  for (int kt = 0; kt < nk; ++kt) {
    const int cur = kt & 1;
    if (kt + 1 < nk) {                    // stage next tile into other buffer
      stageA(cur ^ 1, (kt + 1) * BK);
      stageB(cur ^ 1, (kt + 1) * BK);
    }
    // L2 prefetch two K-tiles ahead (global_prefetch_b8)
    const int pk = (kt + 2) * BK;
    if (pk < Kdim) {
      int gm = rowBase + (tid >> 1);
      if (gm < Mdim)
        __builtin_prefetch(Ab + (long long)gm * lda + pk + ((tid & 1) << 4), 0, 1);
    }

    // ---- fragment loads per ISA 7.12.2 16-bit layouts ----
    FragU a[2], b[4];
#pragma unroll
    for (int t = 0; t < 2; ++t) {
      const __bf16* ap = &Asm[cur][(waveM * 32 + t * 16 + lmod) * LDSA];
#pragma unroll
      for (int j = 0; j < 4; ++j) {
        a[t].u[j]     = *(const unsigned*)(ap + lhalf * 8 + 2 * j);       // K 0..15 half
        a[t].u[j + 4] = *(const unsigned*)(ap + 16 + lhalf * 8 + 2 * j);  // K 16..31 half
      }
    }
#pragma unroll
    for (int s = 0; s < 4; ++s) {
      const __bf16* bp = &Bsm[cur][(waveN * 64 + s * 16 + lmod) * LDSA + lhalf * 8];
#pragma unroll
      for (int j = 0; j < 8; ++j)
        b[s].u[j] = *(const unsigned*)(bp + 2 * (j & 3) + 16 * (j >> 2));
    }

#pragma unroll
    for (int t = 0; t < 2; ++t)
#pragma unroll
      for (int s = 0; s < 4; ++s)
        acc[t][s] = __builtin_amdgcn_wmma_f32_16x16x32_bf16(
            false, a[t].v, false, b[s].v, (short)0, acc[t][s], false, false);

    wait_async_lds();   // staged tile (kt+1) complete before barrier
    __syncthreads();
  }

  // ---- epilogue ----
  const long long cbase = (long long)zq * cso + (long long)zr * csi;
  const float* Rb = (EPI == EPI_BIAS_RES)
                        ? (Rp + (long long)zq * rso + (long long)zr * rsi)
                        : (const float*)nullptr;
#pragma unroll
  for (int t = 0; t < 2; ++t)
#pragma unroll
    for (int s = 0; s < 4; ++s)
#pragma unroll
      for (int j = 0; j < 8; ++j) {
        int gm = rowBase + waveM * 32 + t * 16 + 8 * lhalf + j;
        int gn = colBase + waveN * 64 + s * 16 + lmod;
        if (gm < Mdim && gn < Ndim) {
          float v = acc[t][s][j] * alpha;
          if (EPI >= EPI_BIAS)      v += bias[gn];
          if (EPI == EPI_BIAS_GELU) v = gelu_f(v);
          if (EPI == EPI_BIAS_RES)  v += Rb[(long long)gm * ldc + gn];
          long long off = cbase + (long long)gm * ldc + gn;
          if (OUT_BF16) ((__bf16*)Cp)[off] = (__bf16)v;
          else          ((float*)Cp)[off]  = v;
        }
      }
}

// ---------------------------------------------------------------------------
// Row LayerNorm: f32 in -> bf16 out (one row per block)
// ---------------------------------------------------------------------------
__global__ __launch_bounds__(256) void ln_kernel(
    const float* __restrict__ x, const float* __restrict__ w,
    const float* __restrict__ b, __bf16* __restrict__ out, int C)
{
  const int row = blockIdx.x;
  const float* xr = x + (long long)row * C;
  __shared__ float red[256];

  float s = 0.f;
  for (int i = threadIdx.x; i < C; i += 256) s += xr[i];
  red[threadIdx.x] = s;
  __syncthreads();
  for (int st = 128; st > 0; st >>= 1) {
    if ((int)threadIdx.x < st) red[threadIdx.x] += red[threadIdx.x + st];
    __syncthreads();
  }
  const float mu = red[0] / (float)C;
  __syncthreads();

  s = 0.f;
  for (int i = threadIdx.x; i < C; i += 256) { float d = xr[i] - mu; s += d * d; }
  red[threadIdx.x] = s;
  __syncthreads();
  for (int st = 128; st > 0; st >>= 1) {
    if ((int)threadIdx.x < st) red[threadIdx.x] += red[threadIdx.x + st];
    __syncthreads();
  }
  const float rstd = rsqrtf(red[0] / (float)C + 1e-5f);

  __bf16* orow = out + (long long)row * C;
  for (int i = threadIdx.x; i < C; i += 256)
    orow[i] = (__bf16)((xr[i] - mu) * rstd * w[i] + b[i]);
}

// ---------------------------------------------------------------------------
// Row softmax: f32 scores in -> bf16 probs out (one row per block)
// ---------------------------------------------------------------------------
__global__ __launch_bounds__(256) void softmax_kernel(
    const float* __restrict__ s, __bf16* __restrict__ p, int NC)
{
  const long long row = blockIdx.x;
  const float* sr = s + row * NC;
  __bf16* pr = p + row * NC;
  __shared__ float red[256];

  float mx = -3.402823466e38f;
  for (int i = threadIdx.x; i < NC; i += 256) mx = fmaxf(mx, sr[i]);
  red[threadIdx.x] = mx;
  __syncthreads();
  for (int st = 128; st > 0; st >>= 1) {
    if ((int)threadIdx.x < st)
      red[threadIdx.x] = fmaxf(red[threadIdx.x], red[threadIdx.x + st]);
    __syncthreads();
  }
  const float rowmax = red[0];
  __syncthreads();

  float sum = 0.f;
  for (int i = threadIdx.x; i < NC; i += 256) sum += __expf(sr[i] - rowmax);
  red[threadIdx.x] = sum;
  __syncthreads();
  for (int st = 128; st > 0; st >>= 1) {
    if ((int)threadIdx.x < st) red[threadIdx.x] += red[threadIdx.x + st];
    __syncthreads();
  }
  const float inv = 1.0f / red[0];

  for (int i = threadIdx.x; i < NC; i += 256)
    pr[i] = (__bf16)(__expf(sr[i] - rowmax) * inv);
}

// ---------------------------------------------------------------------------
// Host-side orchestration
// ---------------------------------------------------------------------------
static inline dim3 ggrid(int Mdim, int Ndim, int batch) {
  return dim3((unsigned)((Ndim + BN - 1) / BN),
              (unsigned)((Mdim + BM - 1) / BM),
              (unsigned)batch);
}

struct BlockW {
  const float *ln1w, *ln1b, *qw, *qb, *kw, *kb, *vw, *vb;
  const float *pw, *pb, *ln2w, *ln2b, *f1w, *f1b, *f2w, *f2b;
};

static void run_block(const BlockW& w, float* xres, __bf16* h,
                      __bf16* q, __bf16* k, __bf16* v, __bf16* o, __bf16* mlp,
                      float* scores, __bf16* probs,
                      int C, int Hh, int Ff, hipStream_t stream)
{
  const int hd = C / Hh;
  const int Z  = BB * Hh;
  const float scale = 1.0f / sqrtf((float)hd);
  const long long NN = (long long)NT * NT;

  // h = LN1(x)
  ln_kernel<<<MT, 256, 0, stream>>>(xres, w.ln1w, w.ln1b, h, C);
  // Q/K/V = h @ W + b  (bf16 out)
  gemm_wmma_kernel<true, false, EPI_BIAS, true><<<ggrid(MT, C, 1), 256, 0, stream>>>(
      h, 0, 0, C, w.qw, 0, 0, C, q, 0, 0, C, w.qb, nullptr, 0, 0, 1, MT, C, C, 1.0f);
  gemm_wmma_kernel<true, false, EPI_BIAS, true><<<ggrid(MT, C, 1), 256, 0, stream>>>(
      h, 0, 0, C, w.kw, 0, 0, C, k, 0, 0, C, w.kb, nullptr, 0, 0, 1, MT, C, C, 1.0f);
  gemm_wmma_kernel<true, false, EPI_BIAS, true><<<ggrid(MT, C, 1), 256, 0, stream>>>(
      h, 0, 0, C, w.vw, 0, 0, C, v, 0, 0, C, w.vb, nullptr, 0, 0, 1, MT, C, C, 1.0f);
  // S[z] = scale * Q[z] K[z]^T     z = b*Hh + head
  gemm_wmma_kernel<false, true, EPI_NONE, false><<<ggrid(NT, NT, Z), 256, 0, stream>>>(
      q, (long long)NT * C, hd, C,
      k, (long long)NT * C, hd, C,
      scores, (long long)Hh * NN, NN, NT,
      nullptr, nullptr, 0, 0, Hh, NT, NT, hd, scale);
  // P = softmax(S) (bf16)
  softmax_kernel<<<Z * NT, 256, 0, stream>>>(scores, probs, NT);
  // O[z] = P[z] V[z]  (bf16 out, head-interleaved layout)
  gemm_wmma_kernel<false, false, EPI_NONE, true><<<ggrid(NT, hd, Z), 256, 0, stream>>>(
      probs, (long long)Hh * NN, NN, NT,
      v, (long long)NT * C, hd, C,
      o, (long long)NT * C, hd, C,
      nullptr, nullptr, 0, 0, Hh, NT, hd, NT, 1.0f);
  // x = x + O @ proj + b   (f32 residual stream)
  gemm_wmma_kernel<true, false, EPI_BIAS_RES, false><<<ggrid(MT, C, 1), 256, 0, stream>>>(
      o, 0, 0, C, w.pw, 0, 0, C, xres, 0, 0, C, w.pb, xres, 0, 0, 1, MT, C, C, 1.0f);
  // h = LN2(x)
  ln_kernel<<<MT, 256, 0, stream>>>(xres, w.ln2w, w.ln2b, h, C);
  // mlp = gelu(h @ fc1 + b1)
  gemm_wmma_kernel<true, false, EPI_BIAS_GELU, true><<<ggrid(MT, Ff, 1), 256, 0, stream>>>(
      h, 0, 0, C, w.f1w, 0, 0, Ff, mlp, 0, 0, Ff, w.f1b, nullptr, 0, 0, 1, MT, Ff, C, 1.0f);
  // x = x + mlp @ fc2 + b2
  gemm_wmma_kernel<true, false, EPI_BIAS_RES, false><<<ggrid(MT, C, 1), 256, 0, stream>>>(
      mlp, 0, 0, Ff, w.f2w, 0, 0, C, xres, 0, 0, C, w.f2b, xres, 0, 0, 1, MT, C, Ff, 1.0f);
}

static BlockW block_ptrs(const float* const* in, int base, int l, int C, int Ff) {
  BlockW w;
  size_t lc = (size_t)l * C, lcc = (size_t)l * C * C;
  size_t lcf = (size_t)l * C * Ff, lf = (size_t)l * Ff;
  w.ln1w = in[base + 0] + lc;   w.ln1b = in[base + 1] + lc;
  w.qw   = in[base + 2] + lcc;  w.qb   = in[base + 3] + lc;
  w.kw   = in[base + 4] + lcc;  w.kb   = in[base + 5] + lc;
  w.vw   = in[base + 6] + lcc;  w.vb   = in[base + 7] + lc;
  w.pw   = in[base + 8] + lcc;  w.pb   = in[base + 9] + lc;
  w.ln2w = in[base + 10] + lc;  w.ln2b = in[base + 11] + lc;
  w.f1w  = in[base + 12] + lcf; w.f1b  = in[base + 13] + lf;
  w.f2w  = in[base + 14] + lcf; w.f2b  = in[base + 15] + lc;
  return w;
}

extern "C" void kernel_launch(void* const* d_in, const int* in_sizes, int n_in,
                              void* d_out, int out_size, void* d_ws, size_t ws_size,
                              hipStream_t stream)
{
  (void)in_sizes; (void)n_in; (void)out_size; (void)ws_size;
  const float* in[64];
  for (int i = 0; i < 45 && i < n_in; ++i) in[i] = (const float*)d_in[i];

  // ---- workspace carve-out (~240 MB) ----
  char* ws = (char*)d_ws;
  size_t off = 0;
  auto take = [&](size_t bytes) -> char* {
    char* p = ws + off;
    off += (bytes + 255) & ~(size_t)255;
    return p;
  };
  float*  xres   = (float*) take((size_t)MT * DE * 4);          // residual stream
  __bf16* h      = (__bf16*)take((size_t)MT * DE * 2);          // LN output
  __bf16* q      = (__bf16*)take((size_t)MT * DE * 2);
  __bf16* k      = (__bf16*)take((size_t)MT * DE * 2);
  __bf16* v      = (__bf16*)take((size_t)MT * DE * 2);
  __bf16* o      = (__bf16*)take((size_t)MT * DE * 2);
  __bf16* mlp    = (__bf16*)take((size_t)MT * FE * 2);
  float*  scores = (float*) take((size_t)BB * HE * NT * NT * 4);
  __bf16* probs  = (__bf16*)take((size_t)BB * HE * NT * NT * 2);

  // residual stream <- input x
  (void)hipMemcpyAsync(xres, d_in[0], (size_t)MT * DE * 4,
                       hipMemcpyDeviceToDevice, stream);

  // ---- encoder ----
  for (int l = 0; l < LE; ++l) {
    BlockW w = block_ptrs(in, 1, l, DE, FE);
    run_block(w, xres, h, q, k, v, o, mlp, scores, probs, DE, HE, FE, stream);
  }
  // enc_norm -> h (bf16)
  ln_kernel<<<MT, 256, 0, stream>>>(xres, in[17], in[18], h, DE);
  // decoder embed: xres = h @ W + b  (MT x DD, f32)
  gemm_wmma_kernel<true, false, EPI_BIAS, false><<<ggrid(MT, DD, 1), 256, 0, stream>>>(
      h, 0, 0, DE, in[19], 0, 0, DD, xres, 0, 0, DD, in[20],
      nullptr, 0, 0, 1, MT, DD, DE, 1.0f);

  // ---- decoder ----
  for (int l = 0; l < LDN; ++l) {
    BlockW w = block_ptrs(in, 21, l, DD, FD);
    run_block(w, xres, h, q, k, v, o, mlp, scores, probs, DD, HD, FD, stream);
  }
  // dec_norm -> h
  ln_kernel<<<MT, 256, 0, stream>>>(xres, in[37], in[38], h, DD);

  // ---- prediction head ----
  gemm_wmma_kernel<true, false, EPI_BIAS_GELU, true><<<ggrid(MT, DD, 1), 256, 0, stream>>>(
      h, 0, 0, DD, in[39], 0, 0, DD, q, 0, 0, DD, in[40],
      nullptr, 0, 0, 1, MT, DD, DD, 1.0f);
  gemm_wmma_kernel<true, false, EPI_BIAS_GELU, true><<<ggrid(MT, DD, 1), 256, 0, stream>>>(
      q, 0, 0, DD, in[41], 0, 0, DD, k, 0, 0, DD, in[42],
      nullptr, 0, 0, 1, MT, DD, DD, 1.0f);
  gemm_wmma_kernel<true, false, EPI_BIAS, false><<<ggrid(MT, OUTD, 1), 256, 0, stream>>>(
      k, 0, 0, DD, in[43], 0, 0, OUTD, d_out, 0, 0, OUTD, in[44],
      nullptr, 0, 0, 1, MT, OUTD, DD, 1.0f);
}
